// VectorQuantizer_34239479284071
// MI455X (gfx1250) — matrix operand order repaired and checked by reading, not compiled, not run on hardware
//
#include <hip/hip_runtime.h>
#include <hip/hip_bf16.h>

// ---------------------------------------------------------------------------
// VQ-VAE vector quantizer, fused for MI455X (gfx1250, wave32, WMMA).
//   N = 131072 rows, C = 64 features, K = 1024 codes.
// Distance GEMM uses split-precision bf16 WMMA (hi*hi + hi*lo + lo*hi) at the
// bf16 matrix-core rate with ~f32 argmin accuracy. The whole codebook
// (hi+lo bf16 = 256 KB) lives in LDS (320 KB/WGP). z rows are staged with
// gfx1250 async global->LDS copies overlapped with codebook conversion.
// ---------------------------------------------------------------------------

typedef __attribute__((ext_vector_type(16))) __bf16 bf16x16;
typedef __attribute__((ext_vector_type(4)))  __bf16 bf16x4;
typedef __attribute__((ext_vector_type(8)))  float  v8f;

#define NUM_EMB   1024
#define EMB_DIM   64
#define NROWS     131072          // 32*4096
#define ROWS_PER_BLK 128
#define DECAY_F   0.99f
#define ONE_M_D   0.01f
#define EPS_F     1e-05f
#define CCOST     0.25f

// d_out float offsets (tuple concatenated flat in return order)
#define O_ZQ    0                         // 131072*64 = 8388608
#define O_LOSS  8388608                   // 1
#define O_IDX   8388609                   // 131072
#define O_EMB   8519681                   // 65536
#define O_ECS   8585217                   // 1024
#define O_EMAW  8586241                   // 65536

// ws float offsets
#define W_LOSS  0
#define W_NSUM  1

// LDS carve (bytes): zbuf f32[128*64] | ehi bf16[1024*64] | elo bf16[1024*64]
//                    | enorm f32[1024] | idxbuf i32[128] | red f32[256]
#define L_Z     0
#define L_EHI   32768
#define L_ELO   163840
#define L_ENRM  294912
#define L_IDX   299008
#define L_RED   299520
#define L_TOTAL 300544

// ---------------------------------------------------------------------------
// Kernel A: seed output EMA accumulators + zero ws scalars (runs every call;
// harness poisons buffers and replays the graph).
// ---------------------------------------------------------------------------
__global__ void vq_init_kernel(const float* __restrict__ ema_cluster_size,
                               const float* __restrict__ ema_weight,
                               float* __restrict__ out, float* __restrict__ ws) {
    const int i = blockIdx.x * blockDim.x + threadIdx.x;   // 0..65535
    out[O_EMAW + i] = ema_weight[i] * DECAY_F;             // dw accumulates here
    if (i < NUM_EMB) out[O_ECS + i] = ema_cluster_size[i] * DECAY_F; // counts here
    if (i == 0) { ws[W_LOSS] = 0.0f; ws[W_NSUM] = 0.0f; }
}

// ---------------------------------------------------------------------------
// Kernel B: fused distances + argmin + z_q gather + loss + EMA scatter.
// 1024 blocks x 256 threads (8 waves); each block: 128 rows vs all 1024 codes.
// ---------------------------------------------------------------------------
__global__ void __launch_bounds__(256)
vq_main_kernel(const float* __restrict__ z_e,
               const float* __restrict__ embedding,
               float* __restrict__ out, float* __restrict__ ws) {
    extern __shared__ char smem[];
    float*  zbuf  = (float*)(smem + L_Z);      // [128][64] f32
    __bf16* ehi   = (__bf16*)(smem + L_EHI);   // [1024][64] bf16 hi
    __bf16* elo   = (__bf16*)(smem + L_ELO);   // [1024][64] bf16 residual
    float*  enorm = (float*)(smem + L_ENRM);   // [1024]
    int*    idxbuf= (int*)(smem + L_IDX);      // [128]
    float*  red   = (float*)(smem + L_RED);    // [256]

    const int tid  = threadIdx.x;
    const int lane = tid & 31;
    const int wave = tid >> 5;
    const int rowBase = blockIdx.x * ROWS_PER_BLK;

    // ---- async-stage z rows: 32 KB global -> LDS, overlapped with the
    //      codebook conversion below (gfx1250 GLOBAL_LOAD_ASYNC_TO_LDS_B128,
    //      GVS addressing: SGPR64 base + VGPR32 offset, tracked by ASYNCcnt).
    {
        const float* gbase = z_e + (size_t)rowBase * EMB_DIM;
        unsigned ldsbase = (unsigned)(size_t)(void*)zbuf; // low 32 bits = LDS offset
        #pragma unroll
        for (int it = 0; it < 8; ++it) {
            unsigned off  = (unsigned)(tid * 16 + it * 4096);   // byte offset
            unsigned dlds = ldsbase + off;
            asm volatile("global_load_async_to_lds_b128 %0, %1, %2"
                         :: "v"(dlds), "v"(off), "s"(gbase) : "memory");
        }
    }

    // ---- stage codebook as split bf16 + f32 norms in ONE pass over rows ----
    for (int n = tid; n < NUM_EMB; n += 256) {
        const float4* e4 = (const float4*)(embedding + (size_t)n * EMB_DIM);
        bf16x4* hrow = (bf16x4*)(ehi + (size_t)n * EMB_DIM);
        bf16x4* lrow = (bf16x4*)(elo + (size_t)n * EMB_DIM);
        float s = 0.0f;
        #pragma unroll
        for (int f4 = 0; f4 < EMB_DIM / 4; ++f4) {
            float4 x = e4[f4];
            s += x.x * x.x + x.y * x.y + x.z * x.z + x.w * x.w;
            bf16x4 hv, lv;
            hv.x = (__bf16)x.x; lv.x = (__bf16)(x.x - (float)hv.x);
            hv.y = (__bf16)x.y; lv.y = (__bf16)(x.y - (float)hv.y);
            hv.z = (__bf16)x.z; lv.z = (__bf16)(x.z - (float)hv.z);
            hv.w = (__bf16)x.w; lv.w = (__bf16)(x.w - (float)hv.w);
            hrow[f4] = hv; lrow[f4] = lv;
        }
        enorm[n] = s;
    }

    // drain async z copy, then make everything visible block-wide
    asm volatile("s_wait_asynccnt 0x0" ::: "memory");
    __syncthreads();

    // ---- build persistent A fragments (16 rows per wave, split hi/lo) ----
    // A k order used here (contiguous, not AMD's table order): slot s of lane l holds K = s + 16*(l>=16)
    const int m = lane & 15;        // row within wave tile
    const int h = lane >> 4;        // lane half
    const float* zr = &zbuf[(wave * 16 + m) * EMB_DIM];
    bf16x16 ahi[2], alo[2];
    #pragma unroll
    for (int c = 0; c < 2; ++c) {               // K-chunks of 32 (C=64)
        const int k0 = c * 32 + 16 * h;         // slots 0..7
        const int k1 = c * 32 + 16 * h + 8;     // slots 8..15
        #pragma unroll
        for (int s = 0; s < 8; ++s) {
            float x = zr[k0 + s];
            __bf16 xh = (__bf16)x;
            ahi[c][s] = xh; alo[c][s] = (__bf16)(x - (float)xh);
            float y = zr[k1 + s];
            __bf16 yh = (__bf16)y;
            ahi[c][s + 8] = yh; alo[c][s + 8] = (__bf16)(y - (float)yh);
        }
    }

    // ---- sweep 64 column tiles of 16 codes; fused running argmin.
    //      B fragments double-buffered; two WMMA accumulator chains. ----
    // C/D layout: VGPR r -> (M = r + 8*h, N = lane&15)
    const int j = lane & 15;
    float bestd[8]; int besti[8];
    #pragma unroll
    for (int r = 0; r < 8; ++r) { bestd[r] = 3.4e38f; besti[r] = 0; }

    auto loadB = [&](int nt, bf16x16* Bh, bf16x16* Bl) {
        // B k order used here, same contiguous order as A: lane covers column N = lane&15, K base = 16*(lane>>4)
        const int ncol = nt * 16 + j;
        const __bf16* ph = &ehi[(size_t)ncol * EMB_DIM + h * 16];
        const __bf16* pl = &elo[(size_t)ncol * EMB_DIM + h * 16];
        Bh[0] = *(const bf16x16*)(ph);       Bh[1] = *(const bf16x16*)(ph + 32);
        Bl[0] = *(const bf16x16*)(pl);       Bl[1] = *(const bf16x16*)(pl + 32);
    };
    auto compute = [&](int nt, const bf16x16* Bh, const bf16x16* Bl) {
        v8f a1 = {};   // hi*hi chain
        v8f a2 = {};   // cross-term chain (independent -> deeper WMMA pipelining)
        #pragma unroll
        for (int c = 0; c < 2; ++c) {
            a1 = __builtin_amdgcn_wmma_f32_16x16x32_bf16(false, ahi[c], false, Bh[c],
                                                         (short)0, a1, false, false);
            a2 = __builtin_amdgcn_wmma_f32_16x16x32_bf16(false, ahi[c], false, Bl[c],
                                                         (short)0, a2, false, false);
            a2 = __builtin_amdgcn_wmma_f32_16x16x32_bf16(false, alo[c], false, Bh[c],
                                                         (short)0, a2, false, false);
        }
        const float en = enorm[nt * 16 + j];
        #pragma unroll
        for (int r = 0; r < 8; ++r) {
            float d = en - 2.0f * (a1[r] + a2[r]); // ||z||^2 per-row const: drop
            if (d < bestd[r]) { bestd[r] = d; besti[r] = nt * 16 + j; }
        }
    };

    bf16x16 Bh0[2], Bl0[2], Bh1[2], Bl1[2];
    loadB(0, Bh0, Bl0);
    for (int nt = 0; nt < NUM_EMB / 16; nt += 2) {
        loadB(nt + 1, Bh1, Bl1);                    // prefetch odd tile
        compute(nt, Bh0, Bl0);
        loadB((nt + 2) & 63, Bh0, Bl0);             // prefetch next even tile
        compute(nt + 1, Bh1, Bl1);
    }

    // ---- argmin reduce across the 16 lanes of each N-group ----
    #pragma unroll
    for (int r = 0; r < 8; ++r) {
        #pragma unroll
        for (int off = 1; off < 16; off <<= 1) {
            float od = __shfl_xor(bestd[r], off, 32);
            int   oi = __shfl_xor(besti[r], off, 32);
            if (od < bestd[r] || (od == bestd[r] && oi < besti[r])) {
                bestd[r] = od; besti[r] = oi;
            }
        }
        if (j == 0) idxbuf[wave * 16 + 8 * h + r] = besti[r];
    }
    __syncthreads();

    // ---- outputs: indices, z_q (straight-through), loss, EMA scatter ----
    if (tid < ROWS_PER_BLK) {
        int idx = idxbuf[tid];
        out[O_IDX + rowBase + tid] = (float)idx;
        atomicAdd(&out[O_ECS + idx], ONE_M_D);          // counts * (1-decay)
    }

    float lpart = 0.0f;
    const float4* zb4 = (const float4*)zbuf;
    for (int p = 0; p < (ROWS_PER_BLK * EMB_DIM) / (256 * 4); ++p) {  // 8 iters
        const int q4   = tid + p * 256;          // float4 index, coalesced
        const int row  = q4 >> 4;
        const int feat = (q4 & 15) * 4;
        const int idx  = idxbuf[row];
        float4 z = zb4[q4];
        float4 e = *(const float4*)(embedding + (size_t)idx * EMB_DIM + feat);
        float4 st;                                // STE: z + (q - z)
        st.x = z.x + (e.x - z.x); st.y = z.y + (e.y - z.y);
        st.z = z.z + (e.z - z.z); st.w = z.w + (e.w - z.w);
        *(float4*)(out + O_ZQ + (size_t)(rowBase + row) * EMB_DIM + feat) = st;
        float dx = z.x - e.x, dy = z.y - e.y, dz = z.z - e.z, dw = z.w - e.w;
        lpart += dx * dx + dy * dy + dz * dz + dw * dw;
        float* w = out + O_EMAW + idx * EMB_DIM + feat;   // dw*(1-decay)
        atomicAdd(w + 0, z.x * ONE_M_D); atomicAdd(w + 1, z.y * ONE_M_D);
        atomicAdd(w + 2, z.z * ONE_M_D); atomicAdd(w + 3, z.w * ONE_M_D);
    }
    red[tid] = lpart;
    __syncthreads();
    for (int s = 128; s > 0; s >>= 1) {
        if (tid < s) red[tid] += red[tid + s];
        __syncthreads();
    }
    if (tid == 0) atomicAdd(&ws[W_LOSS], red[0]);
}

// ---------------------------------------------------------------------------
// Kernel C: n = sum(ecs_raw)   (4 blocks x 256)
// ---------------------------------------------------------------------------
__global__ void vq_nsum_kernel(const float* __restrict__ out_ro,
                               float* __restrict__ ws) {
    __shared__ float red[256];
    const int tid = threadIdx.x;
    const int g   = blockIdx.x * 256 + tid;
    red[tid] = (g < NUM_EMB) ? out_ro[O_ECS + g] : 0.0f;
    __syncthreads();
    for (int s = 128; s > 0; s >>= 1) {
        if (tid < s) red[tid] += red[tid + s];
        __syncthreads();
    }
    if (tid == 0) atomicAdd(&ws[W_NSUM], red[0]);
}

// ---------------------------------------------------------------------------
// Kernel D: normalize ecs, new_embedding = ema_weight_new / ecs, final loss.
// 256 blocks x 256 threads -> 65536 (k = i>>6, f = i&63).
// ---------------------------------------------------------------------------
__global__ void vq_final_kernel(float* __restrict__ out,
                                const float* __restrict__ ws) {
    const int i = blockIdx.x * 256 + threadIdx.x;
    const int k = i >> 6, f = i & 63;
    const float n   = ws[W_NSUM];
    const float raw = out[O_ECS + k];
    const float ecs = (raw + EPS_F) / (n + NUM_EMB * EPS_F) * n;
    const float emaw = out[O_EMAW + i];          // already decay*old + 0.01*dw
    __syncthreads();                             // all raw reads before rewrite
    out[O_EMB + i] = emaw / ecs;
    if (f == 0) out[O_ECS + k] = ecs;
    if (i == 0) out[O_LOSS] = CCOST * ws[W_LOSS] / (float)((size_t)NROWS * EMB_DIM);
}

extern "C" void kernel_launch(void* const* d_in, const int* in_sizes, int n_in,
                              void* d_out, int out_size, void* d_ws, size_t ws_size,
                              hipStream_t stream) {
    const float* z_e       = (const float*)d_in[0];
    const float* embedding = (const float*)d_in[1];
    const float* ema_cs    = (const float*)d_in[2];
    const float* ema_w     = (const float*)d_in[3];
    float* out = (float*)d_out;
    float* ws  = (float*)d_ws;
    (void)in_sizes; (void)n_in; (void)out_size; (void)ws_size;

    vq_init_kernel<<<256, 256, 0, stream>>>(ema_cs, ema_w, out, ws);
    vq_main_kernel<<<NROWS / ROWS_PER_BLK, 256, L_TOTAL, stream>>>(z_e, embedding, out, ws);
    vq_nsum_kernel<<<NUM_EMB / 256, 256, 0, stream>>>(out, ws);
    vq_final_kernel<<<(NUM_EMB * EMB_DIM) / 256, 256, 0, stream>>>(out, ws);
}
